// rbm_trans_26362509262865
// MI455X (gfx1250) — matrix operand order repaired and checked by reading, not compiled, not run on hardware
//
#include <hip/hip_runtime.h>
#include <hip/hip_bf16.h>

typedef __attribute__((ext_vector_type(16))) _Float16 v16h;
typedef __attribute__((ext_vector_type(8)))  float    v8f;
typedef __attribute__((ext_vector_type(4)))  unsigned int v4u;
typedef __attribute__((ext_vector_type(8)))  int      v8i;
typedef __attribute__((ext_vector_type(4)))  int      v4i;

#define BB 512
#define NN 128
#define TT 128
#define HH 256
#define LN2F 0.6931471805599453f

// ---------------------------------------------------------------------------
// Kernel 0: convert W (N x H, f32, row-major) -> Wt (H x N, f16, row-major)
// so that a WMMA B-fragment (K contiguous per lane) is one aligned 32B load.
// ---------------------------------------------------------------------------
__global__ void wconv_kernel(const float* __restrict__ W, _Float16* __restrict__ Wt) {
    int i = blockIdx.x * blockDim.x + threadIdx.x;   // 0 .. N*H-1
    int n = i >> 8;          // row of W   (0..127)
    int h = i & 255;         // col of W   (0..255)
    Wt[h * NN + n] = (_Float16)W[i];
}

// ---------------------------------------------------------------------------
// Main fused kernel: one workgroup (256 threads = 8 waves) per batch row.
// Wt (64KB) is staged into LDS once per workgroup by the Tensor Data Mover;
// all 8 waves then consume B-fragments from LDS instead of each re-reading
// 64KB from L2 (8x traffic reduction).
// ---------------------------------------------------------------------------
__global__ void __launch_bounds__(256, 2)
rbm_trans_kernel(const float* __restrict__ x,
                 const float* __restrict__ bias,
                 const float* __restrict__ v,
                 const _Float16* __restrict__ Wt,   // (H x N) f16, in d_ws
                 float* __restrict__ out) {
    __shared__ __align__(32) _Float16 wtLds[HH * NN];  // 64 KB staged W^T
    __shared__ _Float16 xdup[2 * NN];   // duplicated f16 copy of x[b]
    __shared__ float    biasLds[HH];
    __shared__ float    vLds[NN];
    __shared__ float    sumLds[TT];     // sum_h log_cosh per t
    __shared__ float    vdotLds[TT];    // xt . v per t

    const int tid  = threadIdx.x;
    const int lane = tid & 31;
    const int wave = tid >> 5;          // 0..7  -> t-tile row tm = wave
    const int b    = blockIdx.x;

    // ---- TDM: DMA Wt (2D tensor 256 x 128 halfs) into LDS (wave 0 only) ----
    if (wave == 0) {
        const unsigned long long ga = (unsigned long long)(uintptr_t)Wt;
        const unsigned int ldsa     = (unsigned int)(uintptr_t)&wtLds[0];
        // D# group 0: count=1 | lds_addr | global_addr[56:0] | type=2
        v4u g0 = { 1u,
                   ldsa,
                   (unsigned int)(ga & 0xFFFFFFFFu),
                   (unsigned int)((ga >> 32) & 0x01FFFFFFu) | 0x80000000u };
        // D# group 1: data_size=2B; tensor_dim0=128; tensor_dim1=256;
        //             tile_dim0=128; tile_dim1=256; tensor_dim0_stride=128
        v8i g1 = { (int)0x00010000u,          // wg_mask=0, data_size=1 (2B)
                   (int)(128u << 16),         // tensor_dim0[15:0] @ bits63:48
                   (int)(256u << 16),         // tensor_dim1[15:0] @ bits95:80
                   (int)(128u << 16),         // tile_dim0 @ bits127:112
                   (int)0x00000100u,          // tile_dim1=256 @ bits143:128
                   128,                       // tensor_dim0_stride @ bits191:160
                   0, 0 };
        v4i gz = { 0, 0, 0, 0 };
#if defined(__clang_major__) && (__clang_major__ >= 23)
        v8i gz8 = { 0, 0, 0, 0, 0, 0, 0, 0 };
        __builtin_amdgcn_tensor_load_to_lds(g0, g1, gz, gz, gz8, 0);
#else
        __builtin_amdgcn_tensor_load_to_lds(g0, g1, gz, gz, 0);
#endif
    }

    // ---- stage x (duplicated, f16), bias, v ----
    if (tid < NN) {
        _Float16 xv = (_Float16)x[b * NN + tid];
        xdup[tid]      = xv;
        xdup[tid + NN] = xv;
        vLds[tid] = v[tid];
    }
    if (tid < HH) biasLds[tid] = bias[tid];

    if (wave == 0) __builtin_amdgcn_s_wait_tensorcnt(0);  // TDM complete
    __syncthreads();

    // ---- small linear term: vdot[t] = sum_n x[b][(t+n)%128] * v[n] ----
    if (tid < TT) {
        float s = 0.f;
        #pragma unroll 4
        for (int n = 0; n < NN; ++n)
            s += (float)xdup[tid + n] * vLds[n];
        vdotLds[tid] = s;
    }

    // ---- build the 4 A fragments (one per K-step of 32) for this wave ----
    // A fragment k order used here (contiguous, same as B): lanes 0-15: K = 0-15 ; lanes 16-31: K = 16-31
    const int m      = lane & 15;           // row within tile
    const int trow   = wave * 16 + m;       // t index for this lane's A row
    const int kshift = (lane < 16) ? 0 : 16;

    v16h afrag[4];
    #pragma unroll
    for (int ks = 0; ks < 4; ++ks) {
        const int klo = ks * 32 + kshift;
        v16h a;
        #pragma unroll
        for (int e = 0; e < 8; ++e) a[e]     = xdup[trow + klo + e];
        #pragma unroll
        for (int e = 0; e < 8; ++e) a[8 + e] = xdup[trow + klo + 8 + e];
        afrag[ks] = a;
    }

    // ---- GEMM over 16 h-tiles, fused bias + log_cosh + h-accumulation ----
    const int hcol   = lane & 15;                 // column within h-tile
    const int kbshft = (lane < 16) ? 0 : 16;      // B layout: K contiguous per lane

    float rowsum[8];
    #pragma unroll
    for (int r = 0; r < 8; ++r) rowsum[r] = 0.f;

    #pragma unroll 2
    for (int th = 0; th < 16; ++th) {
        const int h = th * 16 + hcol;
        v8f c = {0.f, 0.f, 0.f, 0.f, 0.f, 0.f, 0.f, 0.f};
        #pragma unroll
        for (int ks = 0; ks < 4; ++ks) {
            const int kb = ks * 32 + kbshft;
            // LDS, 32B-aligned: h*128 halfs = 256B, kb*2 bytes multiple of 32
            v16h bf = *(const v16h*)&wtLds[h * NN + kb];
            c = __builtin_amdgcn_wmma_f32_16x16x32_f16(
                    false, afrag[ks], false, bf, (short)0, c, false, false);
        }
        const float bh = biasLds[h];
        #pragma unroll
        for (int r = 0; r < 8; ++r) {
            // log_cosh(p) = |p| + log(1 + exp(-2|p|)) - ln2   (ln2 hoisted out)
            float a = fabsf(c[r] + bh);
            rowsum[r] += a + __logf(1.0f + __expf(-2.0f * a));
        }
    }

    // ---- reduce rowsum over the 16 lanes sharing a t-row, write sumLds ----
    // C layout: VGPR r -> M=r (lanes 0-15), M=r+8 (lanes 16-31); N = lane%16
    #pragma unroll
    for (int r = 0; r < 8; ++r) {
        float s = rowsum[r];
        s += __shfl_xor(s, 1, 16);
        s += __shfl_xor(s, 2, 16);
        s += __shfl_xor(s, 4, 16);
        s += __shfl_xor(s, 8, 16);
        if ((lane & 15) == 0) {
            int t = wave * 16 + r + ((lane >= 16) ? 8 : 0);
            sumLds[t] = s - (float)HH * LN2F;   // apply hoisted -H*ln2 once
        }
    }
    __syncthreads();

    // ---- vals[t] then logsumexp over t (single wave) ----
    if (tid < TT) vdotLds[tid] += sumLds[tid];
    __syncthreads();

    if (tid < 32) {
        float v0 = vdotLds[tid];
        float v1 = vdotLds[tid + 32];
        float v2 = vdotLds[tid + 64];
        float v3 = vdotLds[tid + 96];
        float mx = fmaxf(fmaxf(v0, v1), fmaxf(v2, v3));
        #pragma unroll
        for (int msk = 16; msk >= 1; msk >>= 1)
            mx = fmaxf(mx, __shfl_xor(mx, msk, 32));
        float s = __expf(v0 - mx) + __expf(v1 - mx) + __expf(v2 - mx) + __expf(v3 - mx);
        #pragma unroll
        for (int msk = 16; msk >= 1; msk >>= 1)
            s += __shfl_xor(s, msk, 32);
        if (tid == 0) out[b] = __logf(s) + mx;
    }
}

// ---------------------------------------------------------------------------
extern "C" void kernel_launch(void* const* d_in, const int* in_sizes, int n_in,
                              void* d_out, int out_size, void* d_ws, size_t ws_size,
                              hipStream_t stream) {
    const float* x    = (const float*)d_in[0];   // (B, N)
    const float* W    = (const float*)d_in[1];   // (N, H)
    const float* bias = (const float*)d_in[2];   // (H,)
    const float* v    = (const float*)d_in[3];   // (N,)
    // d_in[4] = translations, structure (n+t)%N exploited directly.

    _Float16* Wt = (_Float16*)d_ws;              // 256*128 f16 = 64 KB

    wconv_kernel<<<(NN * HH) / 256, 256, 0, stream>>>(W, Wt);
    rbm_trans_kernel<<<BB, 256, 0, stream>>>(x, bias, v, Wt, (float*)d_out);
}
